// FastRCNN_29858612641852
// MI455X (gfx1250) — matrix operand order repaired and checked by reading, not compiled, not run on hardware
//
#include <hip/hip_runtime.h>
#include <hip/hip_bf16.h>

// ---------------------------------------------------------------------------
// Fast R-CNN head with relation modules, CDNA5 (gfx1250, wave32, WMMA).
// Big GEMMs: v_wmma_f32_16x16x32_f16; fp32 weights streamed once from HBM and
// converted to f16 in-register (bandwidth roofline: 950MB weights @23.3TB/s).
// Activations pre-packed in the WMMA A-fragment lane layout; A fragments are
// staged in LDS once per block (8 waves share them) to cut L2 traffic 8x.
// ---------------------------------------------------------------------------

#define N_ROI   128
#define CCH     512
#define HF      38
#define WFH     38
#define D_POOL  25088      // 512*49
#define D_FC    4096
#define N_HEADS 16
#define D_HEAD  256
#define ACHUNK  8          // K-steps of A staged in LDS per block (64KB)

typedef __attribute__((ext_vector_type(16))) _Float16 v16h;
typedef __attribute__((ext_vector_type(8)))  float    v8f;

// ---------------------------------------------------------------------------
// WMMA 16-bit A-matrix (16x32) lane layout (ISA 7.12.2):
//   lane L: M = L%16.  lanes 0-15 hold K pairs {0,1},{2,3},{4,5},{6,7},
//   {16,17},{18,19},{20,21},{22,23}; lanes 16-31 hold the +8 shifted pairs.
// Packed buffer: u32[mtile][kstep][lane][8], each u32 = 2 f16 (lo = even K).
// ---------------------------------------------------------------------------
__device__ __forceinline__ size_t apack_half_idx(int n, int k, int K) {
    int mt  = n >> 4, row = n & 15;
    int ks  = k >> 5, kk  = k & 31;
    int grp = ((kk & 15) < 8) ? 0 : 1;
    int lane = row + (grp << 4);
    int i = ((kk & 7) >> 1) + ((kk >> 4) << 2);
    size_t dw = (((size_t)mt * (size_t)(K >> 5) + (size_t)ks) * 32 + lane) * 8 + i;
    return dw * 2 + (kk & 1);
}

// ---------------------------------------------------------------------------
// 1) RoI max pooling (Chainer-style), writes f16 directly in packed A layout.
// ---------------------------------------------------------------------------
__global__ void roi_pool_pack_kernel(const float* __restrict__ x,
                                     const float* __restrict__ rois,
                                     const int*   __restrict__ ridx,
                                     _Float16*    __restrict__ pool_pack) {
    int t = blockIdx.x * blockDim.x + threadIdx.x;   // exact grid
    int n   = t / (CCH * 49);
    int rem = t % (CCH * 49);
    int c   = rem / 49;
    int ph  = (rem % 49) / 7;
    int pw  = rem % 7;

    // xy = (x1,y1,x2,y2) = (rois[:,1], rois[:,0], rois[:,3], rois[:,2])
    float x1 = rois[n * 4 + 1], y1 = rois[n * 4 + 0];
    float x2 = rois[n * 4 + 3], y2 = rois[n * 4 + 2];
    int xmin = (int)rintf(x1 * 0.0625f);
    int ymin = (int)rintf(y1 * 0.0625f);
    int xmax = (int)rintf(x2 * 0.0625f);
    int ymax = (int)rintf(y2 * 0.0625f);
    float roi_w = fmaxf((float)(xmax - xmin + 1), 1.0f);
    float roi_h = fmaxf((float)(ymax - ymin + 1), 1.0f);

    int hs = min(max((int)floorf((float)ph * (roi_h / 7.0f)) + ymin, 0), HF);
    int he = min(max((int)ceilf(((float)ph + 1.0f) * (roi_h / 7.0f)) + ymin, 0), HF);
    int ws = min(max((int)floorf((float)pw * (roi_w / 7.0f)) + xmin, 0), WFH);
    int we = min(max((int)ceilf(((float)pw + 1.0f) * (roi_w / 7.0f)) + xmin, 0), WFH);

    const float* feat = x + ((size_t)ridx[n] * CCH + c) * (HF * WFH);
    float m = -INFINITY;
    for (int h = hs; h < he; ++h)
        for (int w = ws; w < we; ++w)
            m = fmaxf(m, feat[h * WFH + w]);
    if (!(m > -3.0e38f)) m = 0.0f;   // empty bin -> 0

    int k = c * 49 + ph * 7 + pw;
    pool_pack[apack_half_idx(n, k, D_POOL)] = (_Float16)m;
}

// ---------------------------------------------------------------------------
// 2) Main GEMM: partial[split][128][N] = Apack(f16,[128,K]) @ W(f32,[N,K])^T
//    Block = 8 waves, same K-split, consecutive N-tiles. A fragments for a
//    chunk of ACHUNK K-steps are cooperatively staged in LDS (shared by all
//    8 waves); weights are per-wave streamed from HBM with prefetch, and
//    converted f32->f16 in-register.
// ---------------------------------------------------------------------------
union AFrag { int4 q[2]; v16h h; };

__global__ void gemm_wmma_kernel(const int*   __restrict__ Apack,
                                 const float* __restrict__ W,
                                 float*       __restrict__ partial,
                                 int N, int K, int ksteps_per_split) {
    __shared__ int4 As[8 * ACHUNK * 32 * 2];     // [mtile][ksl][lane][2] = 64KB

    int tid  = threadIdx.x;
    int lane = tid & 31;
    int wave = (blockIdx.x * blockDim.x + tid) >> 5;
    int ntiles = N >> 4;
    int ntile = wave % ntiles;
    int split = wave / ntiles;                    // uniform within a block

    int col   = lane & 15;          // B column (N within tile) / C column
    int khalf = (lane >> 4) << 3;   // 0 for lanes 0-15, 8 for lanes 16-31
    int kst_total = K >> 5;

    const float* wrow = W + (size_t)(ntile * 16 + col) * (size_t)K;
    const int4*  Ap4  = (const int4*)Apack;

    v8f acc[8];
#pragma unroll
    for (int m = 0; m < 8; ++m)
#pragma unroll
        for (int j = 0; j < 8; ++j) acc[m][j] = 0.0f;

    int ks_begin = split * ksteps_per_split;
    int ks_end   = ks_begin + ksteps_per_split;

    for (int ks0 = ks_begin; ks0 < ks_end; ks0 += ACHUNK) {
        int len = min(ACHUNK, ks_end - ks0);

        // ---- stage A chunk: 8 mtiles x len ksteps x 1KB -> LDS
        __syncthreads();                          // previous chunk consumed
        if (tid < len * 32) {
#pragma unroll
            for (int m = 0; m < 8; ++m) {
                size_t g = ((size_t)m * kst_total + ks0) * 64 + (size_t)tid * 2;
                int4 a0 = Ap4[g];
                int4 a1 = Ap4[g + 1];
                As[(m * (ACHUNK * 32) + tid) * 2]     = a0;
                As[(m * (ACHUNK * 32) + tid) * 2 + 1] = a1;
            }
        }
        __syncthreads();                          // chunk visible

        for (int ksl = 0; ksl < len; ++ksl) {
            int ks = ks0 + ksl;
            // ---- B fragment: f32 weights K {khalf..+7, 16+khalf..+7} (4x b128) + prefetch
            const float* wp = wrow + (ks << 5) + khalf;
            __builtin_prefetch(wp + (ACHUNK << 5), 0, 1);
            float wf[16];
            *(float4*)(wf + 0)  = *(const float4*)(wp + 0);
            *(float4*)(wf + 4)  = *(const float4*)(wp + 4);
            *(float4*)(wf + 8)  = *(const float4*)(wp + 16);
            *(float4*)(wf + 12) = *(const float4*)(wp + 20);

            // ---- issue all 8 A-fragment LDS reads up front (b128 pairs)
            AFrag af[8];
#pragma unroll
            for (int m = 0; m < 8; ++m) {
                int li = ((m * ACHUNK + ksl) * 32 + lane) * 2;
                af[m].q[0] = As[li];
                af[m].q[1] = As[li + 1];
            }

            v16h b;
#pragma unroll
            for (int i = 0; i < 16; ++i) b[i] = (_Float16)wf[i];

#pragma unroll
            for (int m = 0; m < 8; ++m)
                acc[m] = __builtin_amdgcn_wmma_f32_16x16x32_f16(
                    false, af[m].h, false, b, (short)0, acc[m], false, false);
        }
    }

    // --- store C fragments: VGPR j -> M = j + (lane>=16 ? 8 : 0), N = lane%16
    float* out = partial + (size_t)split * (size_t)(N_ROI) * (size_t)N;
    int hi = (lane >> 4) * 8;
#pragma unroll
    for (int m = 0; m < 8; ++m)
#pragma unroll
        for (int j = 0; j < 8; ++j)
            out[(size_t)(m * 16 + j + hi) * N + ntile * 16 + col] = acc[m][j];
}

// ---------------------------------------------------------------------------
// 3) GEMM epilogue: deterministic split reduce + bias + (relu) ->
//    f32 output and/or packed-f16 activation for the next GEMM (K = 4096).
// ---------------------------------------------------------------------------
__global__ void epilogue_kernel(const float* __restrict__ partial, int nsplit,
                                const float* __restrict__ bias,
                                float*    __restrict__ out_f32,
                                _Float16* __restrict__ out_pack,
                                int act_relu) {
    int idx = blockIdx.x * blockDim.x + threadIdx.x;    // exact: 128*4096
    int o = idx & (D_FC - 1);
    int n = idx >> 12;
    float s = 0.0f;
    for (int sp = 0; sp < nsplit; ++sp)
        s += partial[(size_t)sp * (N_ROI * D_FC) + idx];
    s += bias[o];
    if (act_relu) s = fmaxf(s, 0.0f);
    if (out_f32)  out_f32[idx] = s;
    if (out_pack) out_pack[apack_half_idx(n, o, D_FC)] = (_Float16)s;
}

// ---------------------------------------------------------------------------
// 4) Positional-geometry gates wg1/wg2 ([128,128]).
// ---------------------------------------------------------------------------
__global__ void wg_kernel(const float* __restrict__ rois,
                          const float* __restrict__ Wg1, const float* __restrict__ bg1,
                          const float* __restrict__ Wg2, const float* __restrict__ bg2,
                          float* __restrict__ wg1, float* __restrict__ wg2) {
    int p = blockIdx.x * blockDim.x + threadIdx.x;      // exact: 16384
    int i = p >> 7, j = p & 127;
    float xi1 = rois[i*4+1], yi1 = rois[i*4+0], xi2 = rois[i*4+3], yi2 = rois[i*4+2];
    float xj1 = rois[j*4+1], yj1 = rois[j*4+0], xj2 = rois[j*4+3], yj2 = rois[j*4+2];
    float cxi = 0.5f*(xi1+xi2), cyi = 0.5f*(yi1+yi2);
    float cxj = 0.5f*(xj1+xj2), cyj = 0.5f*(yj1+yj2);
    float hwi = fabsf(cxi - xi2), hhi = fabsf(cyi - yi2);
    float hwj = fabsf(cxj - xj2), hhj = fabsf(cyj - yj2);
    float v4[4];
    v4[0] = fmaxf(fabsf(cxi - cxj), 1e-3f) / hwi;   // tx
    v4[1] = fmaxf(fabsf(cyi - cyj), 1e-3f) / hhi;   // ty
    v4[2] = hwi / hwj;                              // tw
    v4[3] = hhi / hhj;                              // th
    float s1 = 0.0f, s2 = 0.0f;
    for (int cpt = 0; cpt < 4; ++cpt) {
#pragma unroll
        for (int q = 0; q < 8; ++q) {
            float freq = powf(1000.0f, (float)q * (2.0f / 64.0f));
            float sv = sinf(v4[cpt] / freq);
            int base = cpt * 16 + q * 2;
            s1 += sv * (Wg1[base] + Wg1[base + 1]);
            s2 += sv * (Wg2[base] + Wg2[base + 1]);
        }
    }
    wg1[p] = fmaxf(s1 + bg1[0], 0.0f) + 1.0f;
    wg2[p] = fmaxf(s2 + bg2[0], 0.0f) + 1.0f;
}

// ---------------------------------------------------------------------------
// 5) pwa[h,n,m] = dot(K[n,h*256:], Q[m,h*256:]) / 16
// ---------------------------------------------------------------------------
__global__ void pwa_kernel(const float* __restrict__ Km,
                           const float* __restrict__ Qm,
                           float* __restrict__ pwa) {
    __shared__ float Qs[16][D_HEAD];
    int h  = blockIdx.x >> 3;
    int mt = blockIdx.x & 7;
    int tid = threadIdx.x;
    for (int t = tid; t < 16 * D_HEAD; t += 256) {
        int ml = t >> 8, dk = t & 255;
        Qs[ml][dk] = Qm[(size_t)(mt * 16 + ml) * D_FC + h * D_HEAD + dk];
    }
    __syncthreads();
    int ml = tid & 15;
    int n0 = tid >> 4;
    for (int nn = 0; nn < 8; ++nn) {
        int n = n0 * 8 + nn;
        const float* kr = Km + (size_t)n * D_FC + h * D_HEAD;
        float s = 0.0f;
        for (int dk = 0; dk < D_HEAD; ++dk) s += kr[dk] * Qs[ml][dk];
        pwa[((size_t)h * N_ROI + n) * N_ROI + mt * 16 + ml] = s * (1.0f / 16.0f);
    }
}

// ---------------------------------------------------------------------------
// 6) w = exp(pwa - colmax) * wg / sum_head    (one block per head)
// ---------------------------------------------------------------------------
__global__ void attn_norm_kernel(const float* __restrict__ pwa,
                                 const float* __restrict__ wg,
                                 float* __restrict__ wout) {
    __shared__ float cmax[N_ROI];
    __shared__ float red[256];
    int h = blockIdx.x;
    int tid = threadIdx.x;
    if (tid < N_ROI) {
        float mx = -INFINITY;
        for (int n = 0; n < N_ROI; ++n)
            mx = fmaxf(mx, pwa[((size_t)h * N_ROI + n) * N_ROI + tid]);
        cmax[tid] = mx;
    }
    __syncthreads();
    float ls = 0.0f;
    for (int t = tid; t < N_ROI * N_ROI; t += 256) {
        int m = t & 127;
        float wa = expf(pwa[(size_t)h * (N_ROI * N_ROI) + t] - cmax[m]) * wg[t];
        wout[(size_t)h * (N_ROI * N_ROI) + t] = wa;
        ls += wa;
    }
    red[tid] = ls;
    __syncthreads();
    for (int s = 128; s > 0; s >>= 1) {
        if (tid < s) red[tid] += red[tid + s];
        __syncthreads();
    }
    float inv = 1.0f / red[0];
    for (int t = tid; t < N_ROI * N_ROI; t += 256)
        wout[(size_t)h * (N_ROI * N_ROI) + t] *= inv;
}

// ---------------------------------------------------------------------------
// 7) rel[m, h*256+v] = sum_n w[h,n,m] * V[n, h*256+v];
//    writes packed f16 of (base + rel) for the next GEMM.
// ---------------------------------------------------------------------------
__global__ void rel_kernel(const float* __restrict__ w,
                           const float* __restrict__ V,
                           const float* __restrict__ base,
                           _Float16* __restrict__ out_pack) {
    __shared__ float wsh[16][N_ROI];
    int h  = blockIdx.x >> 3;
    int mt = blockIdx.x & 7;
    int tid = threadIdx.x;
    for (int t = tid; t < 16 * N_ROI; t += 256) {
        int ml = t >> 7, n = t & 127;
        wsh[ml][n] = w[((size_t)h * N_ROI + n) * N_ROI + mt * 16 + ml];
    }
    __syncthreads();
    int v = tid;                       // 0..255
    float acc[16];
#pragma unroll
    for (int i = 0; i < 16; ++i) acc[i] = 0.0f;
    for (int n = 0; n < N_ROI; ++n) {
        float vv = V[(size_t)n * D_FC + h * D_HEAD + v];
#pragma unroll
        for (int i = 0; i < 16; ++i) acc[i] += wsh[i][n] * vv;
    }
    int colo = h * D_HEAD + v;
#pragma unroll
    for (int i = 0; i < 16; ++i) {
        int m = mt * 16 + i;
        float r = acc[i] + base[(size_t)m * D_FC + colo];
        out_pack[apack_half_idx(m, colo, D_FC)] = (_Float16)r;
    }
}

// ---------------------------------------------------------------------------
// 8) Output heads: loc [128,84] then sc [128,21], concatenated in d_out.
// ---------------------------------------------------------------------------
__global__ void heads_kernel(const float* __restrict__ fc,
                             const float* __restrict__ Wloc, const float* __restrict__ bloc,
                             const float* __restrict__ Wsc,  const float* __restrict__ bsc,
                             float* __restrict__ out) {
    int t = blockIdx.x * blockDim.x + threadIdx.x;
    const int nloc = N_ROI * 84;
    const int total = nloc + N_ROI * 21;
    if (t >= total) return;
    int n, j; const float* Wp; const float* bp;
    if (t < nloc) { n = t / 84; j = t % 84; Wp = Wloc; bp = bloc; }
    else { int u = t - nloc; n = u / 21; j = u % 21; Wp = Wsc; bp = bsc; }
    const float* a  = fc + (size_t)n * D_FC;
    const float* wr = Wp + (size_t)j * D_FC;
    float s = bp[j];
    for (int k = 0; k < D_FC; ++k) s += a[k] * wr[k];
    out[t] = s;
}

// ---------------------------------------------------------------------------
// Host orchestration
// ---------------------------------------------------------------------------
extern "C" void kernel_launch(void* const* d_in, const int* in_sizes, int n_in,
                              void* d_out, int out_size, void* d_ws, size_t ws_size,
                              hipStream_t stream) {
    (void)in_sizes; (void)n_in; (void)out_size; (void)ws_size;
    const float* x    = (const float*)d_in[0];
    const float* rois = (const float*)d_in[1];
    const int*   ridx = (const int*)  d_in[2];
    const float* W7   = (const float*)d_in[3];  const float* b7  = (const float*)d_in[4];
    const float* W8   = (const float*)d_in[5];  const float* b8  = (const float*)d_in[6];
    const float* Wg1  = (const float*)d_in[7];  const float* bg1 = (const float*)d_in[8];
    const float* Wg2  = (const float*)d_in[9];  const float* bg2 = (const float*)d_in[10];
    const float* Wk1  = (const float*)d_in[11]; const float* bk1 = (const float*)d_in[12];
    const float* Wq1  = (const float*)d_in[13]; const float* bq1 = (const float*)d_in[14];
    const float* Wv1  = (const float*)d_in[15]; const float* bv1 = (const float*)d_in[16];
    const float* Wk2  = (const float*)d_in[17]; const float* bk2 = (const float*)d_in[18];
    const float* Wq2  = (const float*)d_in[19]; const float* bq2 = (const float*)d_in[20];
    const float* Wv2  = (const float*)d_in[21]; const float* bv2 = (const float*)d_in[22];
    const float* Wc   = (const float*)d_in[23]; const float* bc  = (const float*)d_in[24];
    const float* Wloc = (const float*)d_in[25]; const float* bloc= (const float*)d_in[26];
    const float* Wsc  = (const float*)d_in[27]; const float* bsc = (const float*)d_in[28];

    char* wsb = (char*)d_ws;
    size_t off = 0;
    auto take = [&](size_t bytes) -> void* {
        void* p = wsb + off;
        off += (bytes + 255) & ~(size_t)255;
        return p;
    };

    _Float16* pool_pack = (_Float16*)take((size_t)N_ROI * D_POOL * 2);
    float*    partial   = (float*)   take((size_t)8 * N_ROI * D_FC * 4);
    float*    fc7       = (float*)   take((size_t)N_ROI * D_FC * 4);
    _Float16* fc7p      = (_Float16*)take((size_t)N_ROI * D_FC * 2);
    float*    Kmat      = (float*)   take((size_t)N_ROI * D_FC * 4);
    float*    Qmat      = (float*)   take((size_t)N_ROI * D_FC * 4);
    float*    Vmat      = (float*)   take((size_t)N_ROI * D_FC * 4);
    float*    pwa       = (float*)   take((size_t)N_HEADS * N_ROI * N_ROI * 4);
    float*    wbuf      = (float*)   take((size_t)N_HEADS * N_ROI * N_ROI * 4);
    float*    wg1       = (float*)   take((size_t)N_ROI * N_ROI * 4);
    float*    wg2       = (float*)   take((size_t)N_ROI * N_ROI * 4);
    _Float16* act2p     = (_Float16*)take((size_t)N_ROI * D_FC * 2);  // fc7+rel1
    float*    fc8       = (float*)   take((size_t)N_ROI * D_FC * 4);
    _Float16* fc8p      = (_Float16*)take((size_t)N_ROI * D_FC * 2);
    _Float16* act3p     = (_Float16*)take((size_t)N_ROI * D_FC * 2);  // fc8+rel2
    float*    fccls     = (float*)   take((size_t)N_ROI * D_FC * 4);

    // --- 1: RoI pool -> packed f16 A-matrix [128, 25088]
    {
        int total = N_ROI * CCH * 49;               // 3,211,264 (exact /256)
        roi_pool_pack_kernel<<<total / 256, 256, 0, stream>>>(x, rois, ridx, pool_pack);
    }
    // --- wg1/wg2 gates (independent of GEMMs)
    wg_kernel<<<(N_ROI * N_ROI) / 256, 256, 0, stream>>>(rois, Wg1, bg1, Wg2, bg2, wg1, wg2);

    const int NT = D_FC / 16;                       // 256 N-tiles
    auto big_gemm = [&](const _Float16* apack, const float* W, int K, int nsplit,
                        const float* bias, float* outf, _Float16* outp, int relu) {
        int kps = (K / 32) / nsplit;
        int waves = NT * nsplit;
        gemm_wmma_kernel<<<(waves * 32) / 256, 256, 0, stream>>>(
            (const int*)apack, W, partial, D_FC, K, kps);
        epilogue_kernel<<<(N_ROI * D_FC) / 256, 256, 0, stream>>>(
            partial, nsplit, bias, outf, outp, relu);
    };

    // --- 2: fc7 = relu(pool @ W7^T + b7)
    big_gemm(pool_pack, W7, D_POOL, 8, b7, fc7, fc7p, 1);

    // --- 3: relation module 1
    big_gemm(fc7p, Wk1, D_FC, 4, bk1, Kmat, nullptr, 0);
    big_gemm(fc7p, Wq1, D_FC, 4, bq1, Qmat, nullptr, 0);
    big_gemm(fc7p, Wv1, D_FC, 4, bv1, Vmat, nullptr, 0);
    pwa_kernel<<<N_HEADS * 8, 256, 0, stream>>>(Kmat, Qmat, pwa);
    attn_norm_kernel<<<N_HEADS, 256, 0, stream>>>(pwa, wg1, wbuf);
    rel_kernel<<<N_HEADS * 8, 256, 0, stream>>>(wbuf, Vmat, fc7, act2p);

    // --- 4: fc8 = relu((fc7+rel1) @ W8^T + b8)
    big_gemm(act2p, W8, D_FC, 4, b8, fc8, fc8p, 1);

    // --- 5: relation module 2
    big_gemm(fc8p, Wk2, D_FC, 4, bk2, Kmat, nullptr, 0);
    big_gemm(fc8p, Wq2, D_FC, 4, bq2, Qmat, nullptr, 0);
    big_gemm(fc8p, Wv2, D_FC, 4, bv2, Vmat, nullptr, 0);
    pwa_kernel<<<N_HEADS * 8, 256, 0, stream>>>(Kmat, Qmat, pwa);
    attn_norm_kernel<<<N_HEADS, 256, 0, stream>>>(pwa, wg2, wbuf);
    rel_kernel<<<N_HEADS * 8, 256, 0, stream>>>(wbuf, Vmat, fc8, act3p);

    // --- 6: fc_cls = (fc8+rel2) @ Wc^T + bc
    big_gemm(act3p, Wc, D_FC, 4, bc, fccls, nullptr, 0);

    // --- 7: output heads -> d_out (loc [128*84] then sc [128*21])
    {
        int total = N_ROI * 84 + N_ROI * 21;        // 13440
        heads_kernel<<<(total + 255) / 256, 256, 0, stream>>>(
            fccls, Wloc, bloc, Wsc, bsc, (float*)d_out);
    }
}
